// DTIHarmonic_44143673868856
// MI455X (gfx1250) — matrix operand order repaired and checked by reading, not compiled, not run on hardware
//
#include <hip/hip_runtime.h>
#include <hip/hip_bf16.h>

// Problem constants (from the reference).
#define BB  2
#define NN1 384
#define NN2 384
#define DD  128
#define HH  128
#define NL  3

typedef __bf16 bf16;
typedef bf16  v16bf __attribute__((ext_vector_type(16)));
typedef float v8f   __attribute__((ext_vector_type(8)));

__device__ __forceinline__ float sigf(float x) { return 1.0f / (1.0f + expf(-x)); }

// ---------------------------------------------------------------------------
// bf16 WMMA GEMM, fully K-unrolled, compile-time N, 32x16 C tile per wave:
//   two A fragments (rows tm..tm+15, tm+16..tm+31) share one B fragment,
//   2 x v_wmma_f32_16x16x32_bf16 per K-step. Compile-time N lets the
//   strided B-column loads use immediate IOFFSETs (one clause of
//   global_load_b32) instead of a serialized 64-bit address-add chain.
// Fragment layouts (wave32; M/N/C per CDNA5 ISA 7.12.2, K order contiguous, not the ISA table):
//   A 16x32 bf16 : lane L -> row M=L%16 ; elem e -> K = half*16 + e,
//                  half = L/16.
//   B 32x16 bf16 : lane L -> col N=L%16 ; elem e -> K = half*16 + e.
//   C/D 16x16 f32: lane L -> col N=L%16 ; vgpr v -> row M = v + 8*half.
// ---------------------------------------------------------------------------
template <int KK>
__device__ __forceinline__ v16bf load_a_frag(const float* __restrict__ A,
                                             int row, int kk, int half) {
  const float* arow = A + (long long)row * KK + kk + half * 16;
  const float4 a0 = *(const float4*)(arow + 0);
  const float4 a1 = *(const float4*)(arow + 4);
  const float4 a2 = *(const float4*)(arow + 8);
  const float4 a3 = *(const float4*)(arow + 12);
  v16bf af;
  af[0]  = (bf16)a0.x; af[1]  = (bf16)a0.y; af[2]  = (bf16)a0.z; af[3]  = (bf16)a0.w;
  af[4]  = (bf16)a1.x; af[5]  = (bf16)a1.y; af[6]  = (bf16)a1.z; af[7]  = (bf16)a1.w;
  af[8]  = (bf16)a2.x; af[9]  = (bf16)a2.y; af[10] = (bf16)a2.z; af[11] = (bf16)a2.w;
  af[12] = (bf16)a3.x; af[13] = (bf16)a3.y; af[14] = (bf16)a3.z; af[15] = (bf16)a3.w;
  return af;
}

template <int KK, int NN, bool TRANSB, bool HASBIAS, bool RELU>
__global__ __launch_bounds__(32) void wmma_gemm(
    const float* __restrict__ A, const float* __restrict__ Bm,
    const float* __restrict__ bias, float* __restrict__ C,
    long long sA, long long sB, long long sC) {
  const int z = blockIdx.z;
  A  += (long long)z * sA;
  Bm += (long long)z * sB;
  C  += (long long)z * sC;

  const int tm   = blockIdx.x << 5;   // 32 rows per wave
  const int tn   = blockIdx.y << 4;   // 16 cols per wave
  const int lane = threadIdx.x;
  const int half = lane >> 4;
  const int lr   = lane & 15;

  v8f acc0 = {0.f, 0.f, 0.f, 0.f, 0.f, 0.f, 0.f, 0.f};
  v8f acc1 = {0.f, 0.f, 0.f, 0.f, 0.f, 0.f, 0.f, 0.f};

#pragma unroll
  for (int kk = 0; kk < KK; kk += 32) {
    // ---- B fragment (shared by both A fragments)
    v16bf bfrag;
    if (TRANSB) {  // B element [k][n] = Bm[n*K + k]  (e.g. e = hA @ h^T)
      const float* brow = Bm + (long long)(tn + lr) * KK + kk + half * 16;
      const float4 b0 = *(const float4*)(brow + 0);
      const float4 b1 = *(const float4*)(brow + 4);
      const float4 b2 = *(const float4*)(brow + 8);
      const float4 b3 = *(const float4*)(brow + 12);
      bfrag[0]  = (bf16)b0.x; bfrag[1]  = (bf16)b0.y; bfrag[2]  = (bf16)b0.z; bfrag[3]  = (bf16)b0.w;
      bfrag[4]  = (bf16)b1.x; bfrag[5]  = (bf16)b1.y; bfrag[6]  = (bf16)b1.z; bfrag[7]  = (bf16)b1.w;
      bfrag[8]  = (bf16)b2.x; bfrag[9]  = (bf16)b2.y; bfrag[10] = (bf16)b2.z; bfrag[11] = (bf16)b2.w;
      bfrag[12] = (bf16)b3.x; bfrag[13] = (bf16)b3.y; bfrag[14] = (bf16)b3.z; bfrag[15] = (bf16)b3.w;
    } else {
      // B element [k][n] = Bm[k*N + n]; NN is compile-time so e*NN folds
      // into the 24-bit instruction offset -> one clause of 16 loads.
      const float* bcol = Bm + (long long)(kk + half * 16) * NN + tn + lr;
#pragma unroll
      for (int e = 0; e < 16; ++e) bfrag[e] = (bf16)bcol[e * NN];
    }

    // ---- two A fragments sharing bfrag
    const v16bf af0 = load_a_frag<KK>(A, tm + lr, kk, half);
    const v16bf af1 = load_a_frag<KK>(A, tm + 16 + lr, kk, half);

    acc0 = __builtin_amdgcn_wmma_f32_16x16x32_bf16(false, af0, false, bfrag,
                                                   (short)0, acc0, false, false);
    acc1 = __builtin_amdgcn_wmma_f32_16x16x32_bf16(false, af1, false, bfrag,
                                                   (short)0, acc1, false, false);
  }

  const int n = tn + lr;
  const float bv = HASBIAS ? bias[n] : 0.f;
  float* crow = C + (long long)(tm + (half << 3)) * NN + n;
#pragma unroll
  for (int v = 0; v < 8; ++v) {
    float v0 = acc0[v] + bv;
    float v1 = acc1[v] + bv;
    if (RELU) { v0 = fmaxf(v0, 0.f); v1 = fmaxf(v1, 0.f); }
    crow[v * NN] = v0;            // row tm + v + 8*half
    crow[(v + 16) * NN] = v1;     // row tm + 16 + v + 8*half
  }
}

// ---------------------------------------------------------------------------
// Node embedding: out[node,d] = sum_c x[node,c] * W[c,d]   (K=54, VALU)
// ---------------------------------------------------------------------------
__global__ void node_embed(const float* __restrict__ x, const float* __restrict__ W,
                           float* __restrict__ out, int total) {
  int idx = blockIdx.x * blockDim.x + threadIdx.x;
  if (idx >= total) return;
  int node = idx >> 7;  // /128
  int d    = idx & 127;
  const float* xr = x + (long long)node * 54;
  float acc = 0.f;
#pragma unroll
  for (int c = 0; c < 54; ++c) acc = fmaf(xr[c], W[c * DD + d], acc);
  out[idx] = acc;
}

// ---------------------------------------------------------------------------
// GAT attention: symmetrize e, mask by adj, softmax over axis 1 (rows i),
// multiply by adj. One thread per (b, j) column.
// ---------------------------------------------------------------------------
__global__ void gat_att(const float* __restrict__ e, const float* __restrict__ adj,
                        float* __restrict__ att) {
  int idx = blockIdx.x * blockDim.x + threadIdx.x;
  if (idx >= BB * NN1) return;
  int b = idx / NN1, j = idx % NN1;
  const float* eb = e + (long long)b * NN1 * NN1;
  const float* ab = adj + (long long)b * NN1 * NN1;
  float* ob = att + (long long)b * NN1 * NN1;

  float mx = -3.0e38f;
  for (int i = 0; i < NN1; ++i)
    if (ab[i * NN1 + j] > 0.f)
      mx = fmaxf(mx, eb[i * NN1 + j] + eb[j * NN1 + i]);
  float den = 0.f;
  for (int i = 0; i < NN1; ++i)
    if (ab[i * NN1 + j] > 0.f)
      den += expf(eb[i * NN1 + j] + eb[j * NN1 + i] - mx);
  float inv = 1.f / den;
  for (int i = 0; i < NN1; ++i) {
    float aij = ab[i * NN1 + j];
    ob[i * NN1 + j] =
        (aij > 0.f) ? expf(eb[i * NN1 + j] + eb[j * NN1 + i] - mx) * inv * aij : 0.f;
  }
}

// ---------------------------------------------------------------------------
// Gated residual: coeff = sigmoid([x,hp] . gW + gb); out = c*x + (1-c)*hp
// One 128-thread block per node.
// ---------------------------------------------------------------------------
__global__ __launch_bounds__(DD) void gat_gate(const float* __restrict__ x,
                                               const float* __restrict__ hp,
                                               const float* __restrict__ gW,
                                               const float* __restrict__ gb,
                                               float* __restrict__ out) {
  int node = blockIdx.x;
  int d    = threadIdx.x;
  long long base = (long long)node * DD + d;
  float xv = x[base], hv = hp[base];
  __shared__ float red[DD];
  red[d] = xv * gW[d] + hv * gW[DD + d];
  __syncthreads();
#pragma unroll
  for (int s = DD / 2; s > 0; s >>= 1) {
    if (d < s) red[d] += red[d + s];
    __syncthreads();
  }
  float coeff = sigf(red[0] + gb[0]);
  out[base] = coeff * xv + (1.f - coeff) * hv;
}

// ---------------------------------------------------------------------------
// Output init: zero e_c/e_v accumulators, write e_u = duff^2 * delta_uff.
// ---------------------------------------------------------------------------
__global__ void init_out(float* __restrict__ out, const float* __restrict__ delta_uff,
                         const float* __restrict__ duff_coeff) {
  int b = threadIdx.x;
  if (b < BB) {
    out[b * 4 + 0] = 0.f;
    out[b * 4 + 1] = 0.f;
    float dc = duff_coeff[0];
    out[b * 4 + 2] = dc * dc * delta_uff[b];
  }
}

// ---------------------------------------------------------------------------
// Intercept head: hs = sum_i h1g*valid; relu(hs@W1+b1)@W2 + b2 -> out[b,3]
// ---------------------------------------------------------------------------
__global__ __launch_bounds__(DD) void intercept_kernel(
    const float* __restrict__ h1g, const float* __restrict__ valid1,
    const float* __restrict__ W1, const float* __restrict__ b1,
    const float* __restrict__ W2, const float* __restrict__ b2,
    float* __restrict__ out) {
  int b = blockIdx.x;
  int t = threadIdx.x;
  __shared__ float hs[DD];
  __shared__ float tt[HH];
  const float* hb = h1g + (long long)b * NN1 * DD;
  const float* vb = valid1 + b * NN1;
  float acc = 0.f;
  for (int i = 0; i < NN1; ++i) acc = fmaf(hb[i * DD + t], vb[i], acc);
  hs[t] = acc;
  __syncthreads();
  float a2 = b1[t];
  for (int d = 0; d < DD; ++d) a2 = fmaf(hs[d], W1[d * HH + t], a2);
  tt[t] = fmaxf(a2, 0.f) * W2[t];
  __syncthreads();
#pragma unroll
  for (int st = HH / 2; st > 0; st >>= 1) {
    if (t < st) tt[t] += tt[t + st];
    __syncthreads();
  }
  if (t == 0) out[b * 4 + 3] = tt[0] + b2[0];
}

// ---------------------------------------------------------------------------
// Fused pair-MLP + physics. 16x16 pair tile per 256-thread block.
// p1 already has b1 folded in (bias epilogue of its GEMM).
// sp2 rows padded (+4) so 16-lane accesses hit distinct LDS banks.
// ---------------------------------------------------------------------------
__global__ __launch_bounds__(256) void pair_physics(
    const float* __restrict__ p1, const float* __restrict__ p2,
    const float* __restrict__ W2, const float* __restrict__ b2,
    const float* __restrict__ dmv, const float* __restrict__ charge1,
    const float* __restrict__ charge2, const float* __restrict__ vdw_eps,
    const float* __restrict__ vdw_sig, const float* __restrict__ valid1,
    const float* __restrict__ valid2, const float* __restrict__ nm1,
    const float* __restrict__ nm2, const float* __restrict__ vdw_coeff,
    float* __restrict__ out) {
  const int b  = blockIdx.z;
  const int ti = blockIdx.x << 4, tj = blockIdx.y << 4;
  const int t  = threadIdx.x;
  const int li = t >> 4, lj = t & 15;
  const int gi = ti + li, gj = tj + lj;

  __shared__ float sp1[16][DD];
  __shared__ float sp2[16][DD + 4];
  __shared__ float sw2[HH];

  const long long ND = (long long)BB * NN1 * DD;
  float s[5];
#pragma unroll
  for (int m = 0; m < 5; ++m) {
    const float* p1m = p1 + m * ND + ((long long)b * NN1 + ti) * DD;
    const float* p2m = p2 + m * ND + ((long long)b * NN2 + tj) * DD;
    __syncthreads();  // previous iteration's reads complete
#pragma unroll
    for (int r = 0; r < 8; ++r) {
      int idx = t + (r << 8);
      sp1[idx >> 7][idx & 127] = p1m[idx];
      sp2[idx >> 7][idx & 127] = p2m[idx];
    }
    if (t < HH) sw2[t] = W2[m * HH + t];
    __syncthreads();
    float acc = 0.f;
#pragma unroll 4
    for (int h = 0; h < HH; ++h) {
      float v = sp1[li][h] + sp2[lj][h];
      acc = fmaf(fmaxf(v, 0.f), sw2[h], acc);
    }
    s[m] = acc + b2[m];
  }

  const long long pq = ((long long)b * NN1 + gi) * NN2 + gj;
  const float* dv = dmv + pq * 3;
  float dx = dv[0], dy = dv[1], dz = dv[2];
  float dm = sqrtf(fmaf(dx, dx, fmaf(dy, dy, dz * dz)) + 1e-10f);
  if (dm < 0.5f) dm = 1e10f;
  float logdm = logf(dm);

  // coulomb
  float cA  = sigf(s[0]);
  float cN  = fmaf(sigf(s[1]), 2.f, 1.f);
  float q12 = charge1[b * NN1 + gi] * charge2[b * NN2 + gj];
  float ec  = cA * q12 * expf(-cN * logdm);  // (1/dm)^cN
  ec *= valid1[b * NN1 + gi] * valid2[b * NN2 + gj];
  ec = fminf(fmaxf(ec, -100.f), 100.f);

  // vdw
  float vc  = vdw_coeff[0];
  float vA  = fmaf(sigf(s[2]), 0.6f, 0.7f) * (vc * vc) * vdw_eps[pq];
  float vB  = fmaf(tanhf(s[3]), 0.6f, 0.7f);
  float vN  = fmaf(sigf(s[4]), 2.f, 5.f);
  float dm0 = vdw_sig[pq] * vB;
  if (dm0 < 1e-4f) dm0 = 1.f;
  float r  = expf(vN * (logf(dm0) - logdm));  // (dm0/dm)^vN
  float ev = vA * (r * r - 2.f * r);
  ev *= nm1[b * NN1 + gi] * nm2[b * NN2 + gj];
  ev = fminf(ev, 100.f);

  __shared__ float redc[256];
  __shared__ float redv[256];
  redc[t] = ec;
  redv[t] = ev;
  __syncthreads();
#pragma unroll
  for (int st = 128; st > 0; st >>= 1) {
    if (t < st) {
      redc[t] += redc[t + st];
      redv[t] += redv[t + st];
    }
    __syncthreads();
  }
  if (t == 0) {
    atomicAdd(&out[b * 4 + 0], redc[0]);
    atomicAdd(&out[b * 4 + 1], redv[0]);
  }
}

// ---------------------------------------------------------------------------
extern "C" void kernel_launch(void* const* d_in, const int* in_sizes, int n_in,
                              void* d_out, int out_size, void* d_ws, size_t ws_size,
                              hipStream_t stream) {
  (void)in_sizes; (void)n_in; (void)out_size; (void)ws_size;
  // setup_inputs() dict order
  const float* h1        = (const float*)d_in[0];
  const float* h2        = (const float*)d_in[1];
  const float* adj1      = (const float*)d_in[2];
  const float* dmv       = (const float*)d_in[3];
  const float* charge1   = (const float*)d_in[4];
  const float* charge2   = (const float*)d_in[5];
  const float* vdw_eps   = (const float*)d_in[6];
  const float* vdw_sig   = (const float*)d_in[7];
  const float* delta_uff = (const float*)d_in[8];
  const float* valid1    = (const float*)d_in[9];
  const float* valid2    = (const float*)d_in[10];
  const float* nm1       = (const float*)d_in[11];
  const float* nm2       = (const float*)d_in[12];
  const float* node_W    = (const float*)d_in[13];
  const float* gat_W     = (const float*)d_in[14];
  const float* gat_Wb    = (const float*)d_in[15];
  const float* gat_A     = (const float*)d_in[16];
  const float* gat_gW    = (const float*)d_in[17];
  const float* gat_gb    = (const float*)d_in[18];
  const float* pair_W1   = (const float*)d_in[19];
  const float* pair_b1   = (const float*)d_in[20];
  const float* pair_W2   = (const float*)d_in[21];
  const float* pair_b2   = (const float*)d_in[22];
  const float* vdw_coeff = (const float*)d_in[23];
  const float* duff_coeff= (const float*)d_in[24];
  const float* int_W1    = (const float*)d_in[25];
  const float* int_b1    = (const float*)d_in[26];
  const float* int_W2    = (const float*)d_in[27];
  const float* int_b2    = (const float*)d_in[28];
  float* out = (float*)d_out;

  // workspace layout (floats), ~8.7 MB total
  float* ws = (float*)d_ws;
  const long long ND = (long long)BB * NN1 * DD;       // 98304
  const long long NE = (long long)BB * NN1 * NN1;      // 294912
  const long long S  = (long long)NN1 * DD;            // per-batch stride
  float* h1g_a  = ws;
  float* h1g_b  = h1g_a + ND;
  float* h2g    = h1g_b + ND;
  float* hbuf   = h2g + ND;
  float* hAbuf  = hbuf + ND;
  float* hpbuf  = hAbuf + ND;
  float* ebuf   = hpbuf + ND;
  float* attbuf = ebuf + NE;
  float* p1     = attbuf + NE;
  float* p2     = p1 + 5 * ND;

  // 1) node embeddings
  {
    int tot = BB * NN1 * DD;
    node_embed<<<(tot + 255) / 256, 256, 0, stream>>>(h1, node_W, h1g_a, tot);
    node_embed<<<(tot + 255) / 256, 256, 0, stream>>>(h2, node_W, h2g, tot);
  }

  // 2) GAT layers (ligand only)
  float* cur = h1g_a;
  float* nxt = h1g_b;
  dim3 gemm_nd(NN1 / 32, DD / 16, BB);   // [384,128] outputs
  dim3 gemm_ee(NN1 / 32, NN1 / 16, BB);  // [384,384] outputs
  for (int l = 0; l < NL; ++l) {
    const float* W   = gat_W  + (long long)l * DD * DD;
    const float* Wb  = gat_Wb + (long long)l * DD;
    const float* Am  = gat_A  + (long long)l * DD * DD;
    const float* gWl = gat_gW + (long long)l * 2 * DD;
    const float* gbl = gat_gb + l;
    // h = x @ W + Wb
    wmma_gemm<DD, DD, false, true, false><<<gemm_nd, 32, 0, stream>>>(
        cur, W, Wb, hbuf, S, 0, S);
    // hA = h @ A
    wmma_gemm<DD, DD, false, false, false><<<gemm_nd, 32, 0, stream>>>(
        hbuf, Am, nullptr, hAbuf, S, 0, S);
    // e = hA @ h^T
    wmma_gemm<DD, NN1, true, false, false><<<gemm_ee, 32, 0, stream>>>(
        hAbuf, hbuf, nullptr, ebuf, S, S, (long long)NN1 * NN1);
    // masked column softmax * adj
    gat_att<<<(BB * NN1 + 127) / 128, 128, 0, stream>>>(ebuf, adj1, attbuf);
    // h_prime = relu(att @ h)
    wmma_gemm<NN1, DD, false, false, true><<<gemm_nd, 32, 0, stream>>>(
        attbuf, hbuf, nullptr, hpbuf, (long long)NN1 * NN1, S, S);
    // gated residual
    gat_gate<<<BB * NN1, DD, 0, stream>>>(cur, hpbuf, gWl, gbl, nxt);
    float* tmp = cur; cur = nxt; nxt = tmp;
  }

  // 3) output accumulators + e_u term
  init_out<<<1, 32, 0, stream>>>(out, delta_uff, duff_coeff);

  // 4) pair projections: p1[m] = h1g @ W1[m][:D] + b1[m]; p2[m] = h2g @ W1[m][D:]
  for (int m = 0; m < 5; ++m) {
    const float* W1a = pair_W1 + (long long)m * 2 * DD * HH;  // rows 0..127
    const float* W1b = W1a + (long long)DD * HH;              // rows 128..255
    wmma_gemm<DD, HH, false, true, false><<<gemm_nd, 32, 0, stream>>>(
        cur, W1a, pair_b1 + m * HH, p1 + (long long)m * ND, S, 0, S);
    wmma_gemm<DD, HH, false, false, false><<<gemm_nd, 32, 0, stream>>>(
        h2g, W1b, nullptr, p2 + (long long)m * ND, S, 0, S);
  }

  // 5) learned intercept
  intercept_kernel<<<BB, DD, 0, stream>>>(cur, valid1, int_W1, int_b1, int_W2,
                                          int_b2, out);

  // 6) fused pair-MLP + physics + reduction
  dim3 pg(NN1 / 16, NN2 / 16, BB);
  pair_physics<<<pg, 256, 0, stream>>>(p1, p2, pair_W2, pair_b2, dmv, charge1,
                                       charge2, vdw_eps, vdw_sig, valid1, valid2,
                                       nm1, nm2, vdw_coeff, out);
}
